// MultiHeadAttentionLayer_3633542332940
// MI455X (gfx1250) — matrix operand order repaired and checked by reading, not compiled, not run on hardware
//
#include <hip/hip_runtime.h>
#include <hip/hip_bf16.h>
#include <cstdint>

// ---------------------------------------------------------------------------
// MHA (B=4, S=2048, D=512, H=8, DH=64) for gfx1250.
//  - QKV projection: dense v_wmma_f32_16x16x32_f16 GEMM, 16x64 tile per wave.
//  - Attention: flash-attention, block-shared K/V tiles double-buffered into
//    LDS with GLOBAL_LOAD_ASYNC_TO_LDS_B128 (ASYNCcnt), WMMA from LDS.
// ---------------------------------------------------------------------------

#define ATTN_ASYNC 1   // async-to-LDS staging (set 0 for plain LDS staging)

constexpr int B  = 4;
constexpr int S  = 2048;
constexpr int D  = 512;
constexpr int H  = 8;
constexpr int DH = 64;
constexpr int HD = H * DH;     // 512

typedef __attribute__((ext_vector_type(16))) _Float16 v16h;
typedef __attribute__((ext_vector_type(8)))  _Float16 v8h;
typedef __attribute__((ext_vector_type(8)))  float    v8f;

__device__ inline v8f wmma_f16(v16h a, v16h b, v8f c) {
    // D = A(16x32 f16) * B(32x16 f16) + C(16x16 f32)
    return __builtin_amdgcn_wmma_f32_16x16x32_f16(
        /*neg_a=*/false, a, /*neg_b=*/false, b,
        /*c_mod=*/(short)0, c, /*reuse_a=*/false, /*reuse_b=*/false);
}

__device__ inline v16h cat8(v8h lo, v8h hi) {
    return __builtin_shufflevector(lo, hi, 0,1,2,3,4,5,6,7,8,9,10,11,12,13,14,15);
}

// 16-lane-group reductions (rows of the WMMA C layout live in lanes 0-15 / 16-31)
__device__ inline float grp16_max(float v) {
    for (int off = 1; off < 16; off <<= 1) v = fmaxf(v, __shfl_xor(v, off, 32));
    return v;
}
__device__ inline float grp16_sum(float v) {
    for (int off = 1; off < 16; off <<= 1) v += __shfl_xor(v, off, 32);
    return v;
}

// ---- async copy: 16B global -> LDS through the async unit (ASYNCcnt) ------
__device__ inline void copy16_g2l(const _Float16* __restrict__ g, _Float16* l) {
#if ATTN_ASYNC
    // Generic pointer to __shared__ = {aperture_hi32, lds_offset32}; the LDS
    // aperture maps addr[31:0] directly to the LDS byte offset.
    unsigned loff = (unsigned)(unsigned long long)(uintptr_t)l;
    asm volatile("global_load_async_to_lds_b128 %0, %1, off"
                 :: "v"(loff), "v"(g) : "memory");
#else
    *(v8h*)l = *(const v8h*)g;
#endif
}

__device__ inline void wait_async_le2() {
#if ATTN_ASYNC
#if __has_builtin(__builtin_amdgcn_s_wait_asynccnt)
    __builtin_amdgcn_s_wait_asynccnt(2);
#else
    asm volatile("s_wait_asynccnt 0x2" ::: "memory");
#endif
#endif
}
__device__ inline void wait_async_le0() {
#if ATTN_ASYNC
#if __has_builtin(__builtin_amdgcn_s_wait_asynccnt)
    __builtin_amdgcn_s_wait_asynccnt(0);
#else
    asm volatile("s_wait_asynccnt 0x0" ::: "memory");
#endif
#endif
}

// ------------------------- precision / layout prep -------------------------

__global__ void cvt_f32_to_f16(const float* __restrict__ src,
                               _Float16* __restrict__ dst, int n) {
    int i = blockIdx.x * blockDim.x + threadIdx.x;
    if (i < n) dst[i] = (_Float16)src[i];
}

// W[d][n] (D x HD, f32) -> Wt[n][d] (HD x D, f16)
__global__ void transpose_w(const float* __restrict__ W,
                            _Float16* __restrict__ Wt) {
    int i = blockIdx.x * blockDim.x + threadIdx.x;   // over D*HD
    if (i < D * HD) {
        int d = i / HD, n = i % HD;
        Wt[(size_t)n * D + d] = (_Float16)W[i];
    }
}

// ------------------------------ QKV projection -----------------------------
// Each wave: 16 rows (of B*S) x 64 cols output tile; K-loop over D in steps 32.
// Out mode 0: [B][H][S][DH] (Q, K).  Out mode 1: [B][H][DH][S] (V transposed).
__global__ void qkv_proj_wmma(const _Float16* __restrict__ Xh,  // [B*S][D]
                              const _Float16* __restrict__ Wt,  // [HD][D]
                              const float*    __restrict__ bias,// [HD] f32
                              _Float16* __restrict__ Out,
                              int transposed) {
    const int lane = threadIdx.x & 31;
    const int wave = blockIdx.x * (blockDim.x >> 5) + (threadIdx.x >> 5);
    const int rowTile = wave >> 3;          // 0..511  (B*S/16)
    const int colGrp  = wave & 7;           // 0..7    (HD/64)
    const int row0 = rowTile * 16;
    const int n0   = colGrp * 64;
    const int l16  = lane & 15;
    const int hs   = lane >> 4;             // half-wave select

    v8f z = {};
    v8f acc[4] = {z, z, z, z};

    for (int d0 = 0; d0 < D; d0 += 32) {
        // A fragment: row = row0 + l16; K chunks {0-15} (hs=0) / {16-31}
        const _Float16* arow = Xh + (size_t)(row0 + l16) * D + d0 + hs * 16;
        v16h a = cat8(*(const v8h*)arow, *(const v8h*)(arow + 8));
        #pragma unroll
        for (int t = 0; t < 4; ++t) {
            // B fragment: col N = l16 (+16t), K = 16*hs + e (16 contiguous f16)
            const _Float16* brow =
                Wt + (size_t)(n0 + 16 * t + l16) * D + d0 + hs * 16;
            v16h bb = *(const v16h*)brow;
            acc[t] = wmma_f16(a, bb, acc[t]);
        }
    }

    #pragma unroll
    for (int t = 0; t < 4; ++t) {
        int n  = n0 + 16 * t + l16;          // output column
        float bv = bias[n];
        int hh = n >> 6;                     // head
        int dh = n & 63;
        #pragma unroll
        for (int r = 0; r < 8; ++r) {
            int row = row0 + r + 8 * hs;     // global row in B*S
            int bb  = row >> 11;             // /S
            int ss  = row & (S - 1);
            float val = acc[t][r] + bv;
            if (!transposed)
                Out[(((size_t)bb * H + hh) * S + ss) * DH + dh] = (_Float16)val;
            else
                Out[(((size_t)bb * H + hh) * DH + dh) * S + ss] = (_Float16)val;
        }
    }
}

// ----------------------------- flash attention -----------------------------
// Block = 8 waves sharing one (b,h); each wave owns one 16-row q-tile.
// K (32x64) and V^T (64x32) tiles are double-buffered in LDS via async copies.
__global__ void attention_wmma(const _Float16* __restrict__ Qh, // [B][H][S][DH]
                               const _Float16* __restrict__ Kh, // [B][H][S][DH]
                               const _Float16* __restrict__ Vt, // [B][H][DH][S]
                               const int* __restrict__ mask_q,  // [B][S]
                               const int* __restrict__ mask_k,  // [B][S]
                               float* __restrict__ Out) {       // [B][S][HD]
    __shared__ __align__(64) _Float16 Kbuf[2][32 * 64];  // 2 x 4 KB
    __shared__ __align__(64) _Float16 Vbuf[2][64 * 32];  // 2 x 4 KB
    __shared__ __align__(64) _Float16 Pl[8][16 * 32];    // per-wave P, 8 KB

    const int tid   = threadIdx.x;
    const int lane  = tid & 31;
    const int wslot = tid >> 5;
    const int wave  = blockIdx.x * 8 + wslot;    // 0..4095
    const int qTile = wave & 127;                // S/16
    const int bh    = wave >> 7;                 // uniform within block
    const int b = bh >> 3, h = bh & 7;
    const int q0  = qTile * 16;
    const int l16 = lane & 15;
    const int hs  = lane >> 4;

    const _Float16* Qp = Qh + ((size_t)b * H + h) * S * DH;
    const _Float16* Kp = Kh + ((size_t)b * H + h) * S * DH;
    const _Float16* Vp = Vt + ((size_t)b * H + h) * DH * S;

    // staging assignment: each of 256 threads copies 16B of K and 16B of V
    const int krow = tid >> 3, kchunk = tid & 7;   // K tile: 32 rows x 128B
    const int vrow = tid >> 2, vchunk = tid & 3;   // V tile: 64 rows x 64B

    // Q A-fragments for the full DH=64 (two 32-wide K-steps), loaded once.
    v16h qa[2];
    {
        const _Float16* qrow = Qp + (size_t)(q0 + l16) * DH;
        #pragma unroll
        for (int kk = 0; kk < 2; ++kk) {
            const _Float16* p = qrow + kk * 32 + hs * 16;
            qa[kk] = cat8(*(const v8h*)p, *(const v8h*)(p + 8));
        }
    }

    // Per-row (C-layout r index) query mask values.
    float mqf[8];
    #pragma unroll
    for (int r = 0; r < 8; ++r)
        mqf[r] = (float)mask_q[b * S + q0 + r + 8 * hs];

    float m[8], l[8];
    #pragma unroll
    for (int r = 0; r < 8; ++r) { m[r] = -1e30f; l[r] = 0.0f; }
    v8f z = {};
    v8f ctx[4] = {z, z, z, z};

    const float scale = 0.125f;  // 1/sqrt(64)
    _Float16* P = &Pl[wslot][0];

    // prologue: stage tile 0 into buffer 0
    copy16_g2l(Kp + (size_t)krow * DH + kchunk * 8, &Kbuf[0][krow * 64 + kchunk * 8]);
    copy16_g2l(Vp + (size_t)vrow * S  + vchunk * 8, &Vbuf[0][vrow * 32 + vchunk * 8]);

    constexpr int NIT = S / 32;   // 64
    for (int it = 0; it < NIT; ++it) {
        const int k0  = it * 32;
        const int buf = it & 1;
        const bool hasNext = (it + 1) < NIT;

        if (hasNext) {   // stage tile it+1 into buf^1 (uniform branch)
            const int kn = k0 + 32;
            copy16_g2l(Kp + (size_t)(kn + krow) * DH + kchunk * 8,
                       &Kbuf[buf ^ 1][krow * 64 + kchunk * 8]);
            copy16_g2l(Vp + (size_t)vrow * S + kn + vchunk * 8,
                       &Vbuf[buf ^ 1][vrow * 32 + vchunk * 8]);
            wait_async_le2();   // my 2 copies for tile `it` are done (in-order)
        } else {
            wait_async_le0();
        }
        __syncthreads();        // everyone's tile-`it` copies visible in LDS

        // ---- scores: two 16x16 tiles from LDS K
        v8f sc[2];
        float mkf[2];
        #pragma unroll
        for (int j = 0; j < 2; ++j) {
            int kcl = 16 * j + l16;                    // local key column
            mkf[j] = (float)mask_k[b * S + k0 + kcl];
            v8f s_ = z;
            #pragma unroll
            for (int kk = 0; kk < 2; ++kk) {
                const _Float16* krp = &Kbuf[buf][kcl * 64 + kk * 32 + hs * 16];
                v16h bb = *(const v16h*)krp;           // ds_load 32B
                s_ = wmma_f16(qa[kk], bb, s_);
            }
            sc[j] = s_;
        }

        // ---- mask + scale + online softmax (per C-layout row r)
        float resc[8];
        #pragma unroll
        for (int r = 0; r < 8; ++r) {
            float s0 = sc[0][r] * scale + (1.0f - mqf[r] * mkf[0]) * -10000.0f;
            float s1 = sc[1][r] * scale + (1.0f - mqf[r] * mkf[1]) * -10000.0f;
            float tmax = grp16_max(fmaxf(s0, s1));
            float mn = fmaxf(m[r], tmax);
            float rs = __expf(m[r] - mn);
            float e0 = __expf(s0 - mn);
            float e1 = __expf(s1 - mn);
            float psum = grp16_sum(e0 + e1);
            l[r] = l[r] * rs + psum;
            m[r] = mn;
            resc[r] = rs;
            sc[0][r] = e0;
            sc[1][r] = e1;
        }
        #pragma unroll
        for (int t = 0; t < 4; ++t)
            #pragma unroll
            for (int r = 0; r < 8; ++r)
                ctx[t][r] *= resc[r];

        // ---- P (C layout, f32) -> per-wave LDS row-major f16 [16 q][32 k]
        #pragma unroll
        for (int j = 0; j < 2; ++j)
            #pragma unroll
            for (int r = 0; r < 8; ++r)
                P[(r + 8 * hs) * 32 + 16 * j + l16] = (_Float16)sc[j][r];

        // reload as A fragment (same wave; LDS ops in-order per wave)
        v8h plo = *(const v8h*)&P[l16 * 32 + 16 * hs];
        v8h phi = *(const v8h*)&P[l16 * 32 + 16 * hs + 8];
        v16h pa = cat8(plo, phi);

        // ---- ctx += P (16x32) x V (32x64): 4 N-tiles from LDS V
        #pragma unroll
        for (int t = 0; t < 4; ++t) {
            const _Float16* vrp = &Vbuf[buf][(16 * t + l16) * 32 + hs * 16];
            v16h vb = *(const v16h*)vrp;               // ds_load 32B
            ctx[t] = wmma_f16(pa, vb, ctx[t]);
        }

        __syncthreads();   // all waves done reading buf before it is re-filled
    }

    // ---- normalize and store fp32 output [B][S][H*DH]
    #pragma unroll
    for (int t = 0; t < 4; ++t) {
        int dh = 16 * t + l16;
        #pragma unroll
        for (int r = 0; r < 8; ++r) {
            int s = q0 + r + 8 * hs;
            Out[((size_t)b * S + s) * HD + h * DH + dh] = ctx[t][r] / l[r];
        }
    }
}

// --------------------------------- launcher --------------------------------

extern "C" void kernel_launch(void* const* d_in, const int* in_sizes, int n_in,
                              void* d_out, int out_size, void* d_ws, size_t ws_size,
                              hipStream_t stream) {
    const float* x_q   = (const float*)d_in[0];
    const float* x_kv  = (const float*)d_in[1];
    const int*   maskq = (const int*)d_in[2];
    const int*   maskk = (const int*)d_in[3];
    const float* Wq    = (const float*)d_in[4];
    const float* bq    = (const float*)d_in[5];
    const float* Wk    = (const float*)d_in[6];
    const float* bk    = (const float*)d_in[7];
    const float* Wv    = (const float*)d_in[8];
    const float* bv    = (const float*)d_in[9];
    float* out = (float*)d_out;

    // workspace layout (all offsets multiples of 32B -> aligned vector loads)
    char* ws = (char*)d_ws;
    const size_t xBytes = (size_t)B * S * D * sizeof(_Float16);      // 8 MB
    const size_t wBytes = (size_t)HD * D * sizeof(_Float16);         // 512 KB
    const size_t hBytes = (size_t)B * H * S * DH * sizeof(_Float16); // 8 MB
    _Float16* xh_q  = (_Float16*)ws; ws += xBytes;
    _Float16* xh_kv = (_Float16*)ws; ws += xBytes;
    _Float16* Wt_q  = (_Float16*)ws; ws += wBytes;
    _Float16* Wt_k  = (_Float16*)ws; ws += wBytes;
    _Float16* Wt_v  = (_Float16*)ws; ws += wBytes;
    _Float16* Qh    = (_Float16*)ws; ws += hBytes;
    _Float16* Kh    = (_Float16*)ws; ws += hBytes;
    _Float16* Vth   = (_Float16*)ws; ws += hBytes;   // V transposed [B][H][DH][S]

    const int nx = B * S * D;
    cvt_f32_to_f16<<<(nx + 255) / 256, 256, 0, stream>>>(x_q,  xh_q,  nx);
    cvt_f32_to_f16<<<(nx + 255) / 256, 256, 0, stream>>>(x_kv, xh_kv, nx);

    const int nw = D * HD;
    transpose_w<<<(nw + 255) / 256, 256, 0, stream>>>(Wq, Wt_q);
    transpose_w<<<(nw + 255) / 256, 256, 0, stream>>>(Wk, Wt_k);
    transpose_w<<<(nw + 255) / 256, 256, 0, stream>>>(Wv, Wt_v);

    // 4096 wave-tiles per projection, 8 waves / 256-thread block -> 512 blocks
    qkv_proj_wmma<<<512, 256, 0, stream>>>(xh_q,  Wt_q, bq, Qh,  0);
    qkv_proj_wmma<<<512, 256, 0, stream>>>(xh_kv, Wt_k, bk, Kh,  0);
    qkv_proj_wmma<<<512, 256, 0, stream>>>(xh_kv, Wt_v, bv, Vth, 1);

    // 4096 (b,h,q-tile) waves -> 512 blocks of 8 waves, all 8 share (b,h)
    attention_wmma<<<512, 256, 0, stream>>>(Qh, Kh, Vth, maskq, maskk, out);
}
